// WaveBlock_2310692405894
// MI455X (gfx1250) — matrix operand order repaired and checked by reading, not compiled, not run on hardware
//
#include <hip/hip_runtime.h>
#include <hip/hip_bf16.h>
#include <math.h>

// Problem constants (match reference)
#define BB 8
#define CC 192
#define GG 256
#define EE 384
#define RR 4
#define TT 8192
#define LL 3
#define KK 3

typedef __attribute__((ext_vector_type(16))) _Float16 v16h;
typedef __attribute__((ext_vector_type(8)))  _Float16 v8h;
typedef __attribute__((ext_vector_type(8)))  float    v8f;

// ---------------------------------------------------------------------------
// Branch-free exact GELU: erf via Abramowitz-Stegun 7.1.26 (|err| < 1.5e-7,
// far below the f16 output quantization). One v_rcp + one v_exp + 5 FMAs,
// no EXEC divergence (libm erff expands to a branchy monster on gfx1250).
// ---------------------------------------------------------------------------
__device__ __forceinline__ float gelu_exact(float x) {
    float z  = x * 0.70710678118654752f;
    float az = fabsf(z);
    float t  = __builtin_amdgcn_rcpf(fmaf(0.3275911f, az, 1.0f));
    float p  = fmaf(1.061405429f, t, -1.453152027f);
    p = fmaf(p, t, 1.421413741f);
    p = fmaf(p, t, -0.284496736f);
    p = fmaf(p, t, 0.254829592f);
    p = p * t;
    float e  = __expf(-z * z);          // underflow -> erf saturates to +/-1
    float ea = fmaf(-p, e, 1.0f);
    float er = copysignf(ea, z);
    return 0.5f * x * (1.0f + er);
}

// ---------------------------------------------------------------------------
// Fragment loaders (wave32 WMMA 16x16x32 f16; code's own contiguous K order, same for A and B)
// A (16x32 MxK): lanes 0-15 row M=lane, K 0..15; lanes 16-31 row M=lane-16,
//                K 16..31; 2 consecutive K per VGPR.
// B (32x16 KxN): lanes 0-15 col N=lane, K 0..15; lanes 16-31 col N=lane-16, K 16..31.
// ---------------------------------------------------------------------------
__device__ __forceinline__ v16h load_frag_a(const _Float16* __restrict__ rowk, int lhalf) {
    v8h lo = *(const v8h*)(rowk + (lhalf ? 16 : 0));
    v8h hi = *(const v8h*)(rowk + (lhalf ? 24 : 8));
    return __builtin_shufflevector(lo, hi, 0,1,2,3,4,5,6,7,8,9,10,11,12,13,14,15);
}

__device__ __forceinline__ v16h load_frag_b(const _Float16* __restrict__ tile, int lmod, int lhalf) {
    const _Float16* p = tile + lmod * 32 + lhalf * 16;
    v8h lo = *(const v8h*)(p);
    v8h hi = *(const v8h*)(p + 8);
    return __builtin_shufflevector(lo, hi, 0,1,2,3,4,5,6,7,8,9,10,11,12,13,14,15);
}

// ---------------------------------------------------------------------------
// xbuf = x * mask
// ---------------------------------------------------------------------------
__global__ void init_x(const float* __restrict__ x, const float* __restrict__ mask,
                       float* __restrict__ xb) {
    size_t idx = (size_t)blockIdx.x * 256 + threadIdx.x;
    if (idx >= (size_t)BB * CC * TT) return;
    int t = (int)(idx % TT);
    int b = (int)(idx / ((size_t)CC * TT));
    xb[idx] = x[idx] * mask[(size_t)b * TT + t];
}

// ---------------------------------------------------------------------------
// out[b,j] = dot(g2[b,:], w[j,:]) + bias[j]     (rows = C*R or E*R)
// ---------------------------------------------------------------------------
__global__ void lora_vec(const float* __restrict__ g2, const float* __restrict__ w,
                         const float* __restrict__ bias, float* __restrict__ out, int rows) {
    int idx = blockIdx.x * 256 + threadIdx.x;
    if (idx >= BB * rows) return;
    int b = idx / rows, j = idx % rows;
    const float* gv = g2 + (size_t)b * GG;
    const float* wr = w + (size_t)j * GG;
    float s = bias[j];
    #pragma unroll 4
    for (int g = 0; g < GG; ++g) s += gv[g] * wr[g];
    out[idx] = s;
}

// ---------------------------------------------------------------------------
// W[b,m,k] = main[m,k] + sum_r ain[b,k,r] * aout[b,r,m]   -> f16
// ---------------------------------------------------------------------------
__global__ void build_w(const float* __restrict__ mainw, const float* __restrict__ ain,
                        const float* __restrict__ aout, _Float16* __restrict__ out,
                        int MD, int KD) {
    size_t idx = (size_t)blockIdx.x * 256 + threadIdx.x;
    size_t tot = (size_t)BB * MD * KD;
    if (idx >= tot) return;
    int b   = (int)(idx / ((size_t)MD * KD));
    int rem = (int)(idx % ((size_t)MD * KD));
    int m = rem / KD, k = rem % KD;
    float s = mainw[rem];
    #pragma unroll
    for (int r = 0; r < RR; ++r)
        s += ain[(size_t)b * KD * RR + (size_t)k * RR + r] *
             aout[(size_t)b * RR * MD + (size_t)r * MD + m];
    out[idx] = (_Float16)s;
}

// ---------------------------------------------------------------------------
// Fused causal dilated depthwise conv + channel LayerNorm + mask -> packed f16
// Hpk layout: [B][T/16][C/32][t_in=16][k_in=32]
// One block (256 thr) per (b, 16-wide t tile).
// ---------------------------------------------------------------------------
__global__ void __launch_bounds__(256) dconv_ln_pack(
    const float* __restrict__ x,     // (B,C,T)
    const float* __restrict__ mask,  // (B,T)
    const float* __restrict__ w,     // (C,K) layer slice
    const float* __restrict__ bias,  // (C)
    const float* __restrict__ gamma, // (C)
    const float* __restrict__ beta,  // (C)
    _Float16* __restrict__ hpk,
    int dil) {
    __shared__ float h[CC][17];
    __shared__ float ps[16][16], ps2[16][16];
    __shared__ float mean_s[16], rstd_s[16];

    const int b = blockIdx.y, tb = blockIdx.x, t0 = tb * 16;
    const int tid = threadIdx.x;

    for (int idx = tid; idx < CC * 16; idx += 256) {
        int c = idx >> 4, tt = idx & 15, t = t0 + tt;
        const float* xr = x + ((size_t)b * CC + c) * TT;
        float acc = bias[c];
        #pragma unroll
        for (int k = 0; k < KK; ++k) {
            int ts = t - k * dil;
            float xv = (ts >= 0) ? xr[ts] : 0.0f;
            acc += w[c * KK + k] * xv;
        }
        h[c][tt] = acc;
    }
    __syncthreads();

    { // partial sums over channels: 16 groups of 12
        int tt = tid & 15, grp = tid >> 4;
        float s = 0.f, s2 = 0.f;
        for (int c = grp * 12; c < grp * 12 + 12; ++c) {
            float v = h[c][tt]; s += v; s2 += v * v;
        }
        ps[grp][tt] = s; ps2[grp][tt] = s2;
    }
    __syncthreads();
    if (tid < 16) {
        float s = 0.f, s2 = 0.f;
        #pragma unroll
        for (int g = 0; g < 16; ++g) { s += ps[g][tid]; s2 += ps2[g][tid]; }
        float mu  = s / (float)CC;
        float var = s2 / (float)CC - mu * mu;
        mean_s[tid] = mu;
        rstd_s[tid] = rsqrtf(var + 1e-5f);
    }
    __syncthreads();

    const size_t base = (((size_t)b * (TT / 16) + tb) * (CC / 32)) * 512;
    for (int p = tid; p < CC * 16; p += 256) {
        int kb = p >> 9, t_in = (p >> 5) & 15, k_in = p & 31;
        int c = kb * 32 + k_in;
        float v = (h[c][t_in] - mean_s[t_in]) * rstd_s[t_in] * gamma[c] + beta[c];
        v *= mask[(size_t)b * TT + t0 + t_in];
        hpk[base + p] = (_Float16)v;
    }
}

// ---------------------------------------------------------------------------
// GEMM1: Y = W1_b (E x C) @ H (C x T), + bias, exact GELU -> packed f16 Gpk
// 8 waves (all M), each 3 M-tiles x 4 N-tiles -> block covers 384 x 64.
// Per K-step per wave: 14 VMEM b128 loads feed 12 WMMAs.
// ---------------------------------------------------------------------------
__global__ void __launch_bounds__(256) gemm1_gelu(
    const _Float16* __restrict__ Wh,  // (B,E,C) f16
    const _Float16* __restrict__ Hpk, // (B,T/16,C/32,512)
    const float* __restrict__ bias,   // (E)
    _Float16* __restrict__ Gpk) {     // (B,T/16,E/32,512)
    constexpr int KB = CC / 32; // 6
    constexpr int NT = 4;
    const int b = blockIdx.y, tid = threadIdx.x;
    const int wave = tid >> 5, lane = tid & 31, lmod = lane & 15, lhalf = lane >> 4;
    const int tb0 = blockIdx.x * NT;

    v8f acc[3][NT];
    #pragma unroll
    for (int i = 0; i < 3; ++i)
        #pragma unroll
        for (int j = 0; j < NT; ++j) acc[i][j] = (v8f){0,0,0,0,0,0,0,0};

    const _Float16* wb = Wh + (size_t)b * EE * CC;
    const _Float16* hb = Hpk + ((size_t)b * (TT / 16) + tb0) * ((size_t)KB * 512);

    #pragma unroll 2
    for (int kb = 0; kb < KB; ++kb) {
        v16h a[3], bm[NT];
        #pragma unroll
        for (int mt = 0; mt < 3; ++mt) {
            int mrow = wave * 48 + mt * 16 + lmod;
            a[mt] = load_frag_a(wb + (size_t)mrow * CC + kb * 32, lhalf);
        }
        #pragma unroll
        for (int nt = 0; nt < NT; ++nt)
            bm[nt] = load_frag_b(hb + (size_t)nt * (KB * 512) + (size_t)kb * 512, lmod, lhalf);
        #pragma unroll
        for (int mt = 0; mt < 3; ++mt)
            #pragma unroll
            for (int nt = 0; nt < NT; ++nt)
                acc[mt][nt] = __builtin_amdgcn_wmma_f32_16x16x32_f16(
                    false, a[mt], false, bm[nt], (short)0, acc[mt][nt], false, false);
    }

    #pragma unroll
    for (int mt = 0; mt < 3; ++mt) {
        const int m0 = wave * 48 + mt * 16 + lhalf * 8;
        #pragma unroll
        for (int nt = 0; nt < NT; ++nt) {
            const int tb = tb0 + nt;
            v8h o;
            #pragma unroll
            for (int v = 0; v < 8; ++v) {
                float val = acc[mt][nt][v] + bias[m0 + v];
                o[v] = (_Float16)gelu_exact(val);
            }
            _Float16* dst = Gpk +
                (((size_t)b * (TT / 16) + tb) * (EE / 32) + (m0 >> 5)) * 512 +
                lmod * 32 + (m0 & 31);
            *(v8h*)dst = o;
        }
    }
}

// ---------------------------------------------------------------------------
// GEMM2: Y = W2_b (C x E) @ G (E x T), + bias + residual -> fp32 (B,C,T)
// 8 waves as 4(M) x 2(N); each wave 3 M-tiles x 4 N-tiles -> block 192 x 128.
// ---------------------------------------------------------------------------
__global__ void __launch_bounds__(256) gemm2_res(
    const _Float16* __restrict__ Wh,  // (B,C,E) f16
    const _Float16* __restrict__ Gpk, // (B,T/16,E/32,512)
    const float* __restrict__ bias,   // (C)
    const float* __restrict__ xres,   // (B,C,T)
    float* __restrict__ xout) {       // (B,C,T)
    constexpr int KB = EE / 32; // 12
    constexpr int NT = 4;
    const int b = blockIdx.y, tid = threadIdx.x;
    const int wave = tid >> 5, lane = tid & 31, lmod = lane & 15, lhalf = lane >> 4;
    const int wm = wave & 3, wn = wave >> 2;
    const int tb0 = blockIdx.x * (2 * NT) + wn * NT;

    v8f acc[3][NT];
    #pragma unroll
    for (int i = 0; i < 3; ++i)
        #pragma unroll
        for (int j = 0; j < NT; ++j) acc[i][j] = (v8f){0,0,0,0,0,0,0,0};

    const _Float16* wb = Wh + (size_t)b * CC * EE;
    const _Float16* gb = Gpk + ((size_t)b * (TT / 16) + tb0) * ((size_t)KB * 512);

    #pragma unroll 2
    for (int kb = 0; kb < KB; ++kb) {
        v16h a[3], bm[NT];
        #pragma unroll
        for (int mt = 0; mt < 3; ++mt) {
            int mrow = wm * 48 + mt * 16 + lmod;
            a[mt] = load_frag_a(wb + (size_t)mrow * EE + kb * 32, lhalf);
        }
        #pragma unroll
        for (int nt = 0; nt < NT; ++nt)
            bm[nt] = load_frag_b(gb + (size_t)nt * (KB * 512) + (size_t)kb * 512, lmod, lhalf);
        #pragma unroll
        for (int mt = 0; mt < 3; ++mt)
            #pragma unroll
            for (int nt = 0; nt < NT; ++nt)
                acc[mt][nt] = __builtin_amdgcn_wmma_f32_16x16x32_f16(
                    false, a[mt], false, bm[nt], (short)0, acc[mt][nt], false, false);
    }

    #pragma unroll
    for (int mt = 0; mt < 3; ++mt) {
        const int m0 = wm * 48 + mt * 16 + lhalf * 8;
        #pragma unroll
        for (int nt = 0; nt < NT; ++nt) {
            const int t = (tb0 + nt) * 16 + lmod;
            #pragma unroll
            for (int v = 0; v < 8; ++v) {
                int m = m0 + v;
                size_t idx = ((size_t)b * CC + m) * TT + t;
                xout[idx] = acc[mt][nt][v] + bias[m] + xres[idx];
            }
        }
    }
}

// ---------------------------------------------------------------------------
// Host launcher
// ---------------------------------------------------------------------------
extern "C" void kernel_launch(void* const* d_in, const int* in_sizes, int n_in,
                              void* d_out, int out_size, void* d_ws, size_t ws_size,
                              hipStream_t stream) {
    (void)in_sizes; (void)n_in; (void)out_size; (void)ws_size;
    const float* x        = (const float*)d_in[0];
    const float* x_mask   = (const float*)d_in[1];
    const float* g        = (const float*)d_in[2];   // (B,G,1) == (B,G)
    const float* dconv_w  = (const float*)d_in[3];   // (L,C,K)
    const float* dconv_b  = (const float*)d_in[4];   // (L,C)
    const float* ln_gamma = (const float*)d_in[5];   // (L,C)
    const float* ln_beta  = (const float*)d_in[6];   // (L,C)
    const float* p1_w     = (const float*)d_in[7];   // (L,E,C)
    const float* p1_b     = (const float*)d_in[8];   // (L,E)
    const float* p1_ain_w = (const float*)d_in[9];   // (L,C*R,G)
    const float* p1_ain_b = (const float*)d_in[10];  // (L,C*R)
    const float* p1_aout_w= (const float*)d_in[11];  // (L,E*R,G)
    const float* p1_aout_b= (const float*)d_in[12];  // (L,E*R)
    const float* p2_w     = (const float*)d_in[13];  // (L,C,E)
    const float* p2_b     = (const float*)d_in[14];  // (L,C)
    const float* p2_ain_w = (const float*)d_in[15];  // (L,E*R,G)
    const float* p2_ain_b = (const float*)d_in[16];  // (L,E*R)
    const float* p2_aout_w= (const float*)d_in[17];  // (L,C*R,G)
    const float* p2_aout_b= (const float*)d_in[18];  // (L,C*R)
    float* out = (float*)d_out;

    // Workspace carve-up
    char* ws = (char*)d_ws;
    size_t off = 0;
    auto carve = [&](size_t bytes) -> char* {
        char* p = ws + off;
        off = (off + bytes + 255) & ~(size_t)255;
        return p;
    };
    float*    xbuf  = (float*)   carve((size_t)BB * CC * TT * 4);
    _Float16* hpk   = (_Float16*)carve((size_t)BB * CC * TT * 2);
    _Float16* gpk   = (_Float16*)carve((size_t)BB * EE * TT * 2);
    _Float16* w1h   = (_Float16*)carve((size_t)BB * EE * CC * 2);
    _Float16* w2h   = (_Float16*)carve((size_t)BB * CC * EE * 2);
    float*    a1in  = (float*)   carve((size_t)BB * CC * RR * 4);
    float*    a1out = (float*)   carve((size_t)BB * EE * RR * 4);
    float*    a2in  = (float*)   carve((size_t)BB * EE * RR * 4);
    float*    a2out = (float*)   carve((size_t)BB * CC * RR * 4);

    const int dil[LL] = {1, 3, 9};

    // x = x * mask
    {
        size_t n = (size_t)BB * CC * TT;
        init_x<<<dim3((unsigned)((n + 255) / 256)), dim3(256), 0, stream>>>(x, x_mask, xbuf);
    }

    for (int l = 0; l < LL; ++l) {
        // LoRA vectors: g @ W^T + b
        lora_vec<<<dim3((BB * CC * RR + 255) / 256), dim3(256), 0, stream>>>(
            g, p1_ain_w + (size_t)l * CC * RR * GG, p1_ain_b + (size_t)l * CC * RR, a1in, CC * RR);
        lora_vec<<<dim3((BB * EE * RR + 255) / 256), dim3(256), 0, stream>>>(
            g, p1_aout_w + (size_t)l * EE * RR * GG, p1_aout_b + (size_t)l * EE * RR, a1out, EE * RR);
        lora_vec<<<dim3((BB * EE * RR + 255) / 256), dim3(256), 0, stream>>>(
            g, p2_ain_w + (size_t)l * EE * RR * GG, p2_ain_b + (size_t)l * EE * RR, a2in, EE * RR);
        lora_vec<<<dim3((BB * CC * RR + 255) / 256), dim3(256), 0, stream>>>(
            g, p2_aout_w + (size_t)l * CC * RR * GG, p2_aout_b + (size_t)l * CC * RR, a2out, CC * RR);

        // Effective per-batch weights (fold rank-4 update), f16
        build_w<<<dim3((unsigned)(((size_t)BB * EE * CC + 255) / 256)), dim3(256), 0, stream>>>(
            p1_w + (size_t)l * EE * CC, a1in, a1out, w1h, EE, CC);
        build_w<<<dim3((unsigned)(((size_t)BB * CC * EE + 255) / 256)), dim3(256), 0, stream>>>(
            p2_w + (size_t)l * CC * EE, a2in, a2out, w2h, CC, EE);

        // Conv + LN + mask -> packed f16
        dconv_ln_pack<<<dim3(TT / 16, BB), dim3(256), 0, stream>>>(
            xbuf, x_mask,
            dconv_w + (size_t)l * CC * KK, dconv_b + (size_t)l * CC,
            ln_gamma + (size_t)l * CC, ln_beta + (size_t)l * CC,
            hpk, dil[l]);

        // GEMM1 + bias + GELU -> packed f16 (block covers 384 x 64)
        gemm1_gelu<<<dim3(TT / 64, BB), dim3(256), 0, stream>>>(
            w1h, hpk, p1_b + (size_t)l * EE, gpk);

        // GEMM2 + bias + residual -> fp32 (block covers 192 x 128)
        float* dst = (l == LL - 1) ? out : xbuf;
        gemm2_res<<<dim3(TT / 128, BB), dim3(256), 0, stream>>>(
            w2h, gpk, p2_b + (size_t)l * CC, xbuf, dst);
    }
}
